// SquaredL2Distance_55722905699254
// MI455X (gfx1250) — compile-verified, matrix operand order checked by reading, not run on hardware
//
#include <hip/hip_runtime.h>

// out[n,m] = max(|q_n|^2 + |s_m|^2 - 2 q_n.s_m, 0)
// Cross term: split-precision bf16 WMMA (bf16x3: hi*hi + hi*lo + lo*hi), fp32 accum.
// Compute-bound: 3 * 2*N*M*D = 206 GFLOP of bf16 WMMA vs 256MB output (~11us HBM).
// Path A (preferred): prepass splits fp32 -> tiled bf16 hi/lo in workspace; GEMM
// stages tiles with async global->LDS DMA (ASYNCcnt), zero conversion VALU in loop.
// Path B (small ws): convert in-kernel (cheap trunc/residual + pk conversions).
// Block = 64x128 out tile, 4 wave32s, wave tile 32x64: 24 WMMA : 24 ds_load_b128.

typedef __attribute__((ext_vector_type(16))) __bf16 v16bf;
typedef __attribute__((ext_vector_type(8)))  float  v8f;
typedef __attribute__((ext_vector_type(2)))  __bf16 v2bf;
typedef __attribute__((ext_vector_type(4)))  int    v4i;

#define AS1 __attribute__((address_space(1)))
#define AS3 __attribute__((address_space(3)))

#define D_DIM 512
#define BM 64
#define BN 128
#define BK 32
#define NTHREADS 128
#define NCHUNK (D_DIM / BK)
#define LDST 40                            // halves per LDS row: 32 data + 8 pad
#define QH_OFF 0
#define QL_OFF (BM * LDST)                 // 2560 halves
#define SH_OFF (2 * BM * LDST)             // 5120
#define SL_OFF (2 * BM * LDST + BN * LDST) // 10240
#define BUF_HALVES (2 * BM * LDST + 2 * BN * LDST) // 15360 halves = 30 KB/buffer

#if __has_builtin(__builtin_amdgcn_global_load_async_to_lds_b128)
#define HAVE_ASYNC_LDS 1
#else
#define HAVE_ASYNC_LDS 0
#endif

// addrspace reinterpretation: global generic ptr value == as(1) address;
// low 32 bits of a generic LDS pointer == LDS byte offset (as(3) address).
static __device__ __forceinline__ AS1 v4i* to_global_v4i(const void* p) {
  return (AS1 v4i*)(unsigned long long)p;
}
static __device__ __forceinline__ AS3 v4i* to_lds_v4i(void* p) {
  return (AS3 v4i*)(unsigned int)(unsigned long long)p;
}

static __device__ __forceinline__ void wait_async0() {
#if __has_builtin(__builtin_amdgcn_s_wait_asynccnt)
  __builtin_amdgcn_s_wait_asynccnt(0);
#else
  asm volatile("s_wait_asynccnt 0" ::: "memory");
#endif
}

// ---------------- conversion helpers ----------------

static __device__ __forceinline__ unsigned short f2bf_rn(float f) {
  unsigned int u = __float_as_uint(f);
  u += 0x7FFFu + ((u >> 16) & 1u);
  return (unsigned short)(u >> 16);
}

// pack truncated-bf16 of two fp32 bit images: (u1>>16)<<16 | (u0>>16)  (one v_perm_b32)
static __device__ __forceinline__ unsigned int pack_hi_trunc(unsigned int u0, unsigned int u1) {
  return __builtin_amdgcn_perm(u1, u0, 0x07060302u);
}

static __device__ __forceinline__ unsigned int pack_bf16_rn(float a, float b) {
#if __has_builtin(__builtin_amdgcn_cvt_pk_bf16_f32)
  union { v2bf v; unsigned int u; } c;
  c.v = __builtin_amdgcn_cvt_pk_bf16_f32(a, b);
  return c.u;
#else
  return (unsigned int)f2bf_rn(a) | ((unsigned int)f2bf_rn(b) << 16);
#endif
}

union Frag { v16bf v; uint4 q[2]; };
union Acc  { v8f  v; float f[8]; };

// 16-bit A/B fragment per ISA layout: lane = (row%16) + 16*khalf,
// per-lane halves cover K = khalf*8 + [0..8) and khalf*8 + 16 + [0..8).
static __device__ __forceinline__ v16bf load_frag(const unsigned short* base, int row0, int lane) {
  const int m  = lane & 15;
  const int kh = lane >> 4;
  const unsigned short* p = base + (row0 + m) * LDST + kh * 8;
  Frag f;
  f.q[0] = *(const uint4*)(p);
  f.q[1] = *(const uint4*)(p + 16);
  return f.v;
}

// 24 WMMAs per wave per K-chunk (2 ms x 4 ns x 3 split terms), 24 ds_load_b128.
static __device__ __forceinline__ void compute_chunk(const unsigned short* b, int wm, int wn,
                                                     int lane, v8f acc[2][4]) {
  v16bf aH[2], aL[2], bH[4], bL[4];
#pragma unroll
  for (int ms = 0; ms < 2; ++ms) {
    aH[ms] = load_frag(b + QH_OFF, wm * 32 + ms * 16, lane);
    aL[ms] = load_frag(b + QL_OFF, wm * 32 + ms * 16, lane);
  }
#pragma unroll
  for (int ns = 0; ns < 4; ++ns) {
    bH[ns] = load_frag(b + SH_OFF, wn * 64 + ns * 16, lane);
    bL[ns] = load_frag(b + SL_OFF, wn * 64 + ns * 16, lane);
  }
#pragma unroll
  for (int ms = 0; ms < 2; ++ms)
#pragma unroll
    for (int ns = 0; ns < 4; ++ns) {
      acc[ms][ns] = __builtin_amdgcn_wmma_f32_16x16x32_bf16(
          false, aH[ms], false, bH[ns], (short)0, acc[ms][ns], false, false);
      acc[ms][ns] = __builtin_amdgcn_wmma_f32_16x16x32_bf16(
          false, aH[ms], false, bL[ns], (short)0, acc[ms][ns], false, false);
      acc[ms][ns] = __builtin_amdgcn_wmma_f32_16x16x32_bf16(
          false, aL[ms], false, bH[ns], (short)0, acc[ms][ns], false, false);
    }
}

static __device__ __forceinline__ void epilogue(const v8f acc[2][4], const float* q2,
                                                const float* s2, float* out, int rowBase,
                                                int colBase, int wm, int wn, int lane, int mcols) {
  const int lcol  = lane & 15;
  const int lhalf = lane >> 4;
#pragma unroll
  for (int ms = 0; ms < 2; ++ms)
#pragma unroll
    for (int ns = 0; ns < 4; ++ns) {
      const int row0 = rowBase + wm * 32 + ms * 16;
      const int col0 = colBase + wn * 64 + ns * 16;
      const float s2v = s2[col0 + lcol];
      Acc a; a.v = acc[ms][ns];
#pragma unroll
      for (int r = 0; r < 8; ++r) {
        const int row = row0 + r + 8 * lhalf;
        float d = q2[row] + s2v - 2.0f * a.f[r];
        out[(size_t)row * mcols + col0 + lcol] = fmaxf(d, 0.0f);
      }
    }
}

// ---------------- prep kernels ----------------

// Row sum-of-squares: one wave per row, 8 rows per block.
__global__ __launch_bounds__(256) void rowsumsq_kernel(const float* __restrict__ X,
                                                       float* __restrict__ out) {
  const int lane = threadIdx.x & 31;
  const int wave = threadIdx.x >> 5;
  const int row  = blockIdx.x * 8 + wave;
  const float4* p = (const float4*)(X + (size_t)row * D_DIM);
  float s = 0.0f;
#pragma unroll
  for (int i = 0; i < 4; ++i) {
    float4 v = p[lane + i * 32];
    s += v.x * v.x + v.y * v.y + v.z * v.z + v.w * v.w;
  }
#pragma unroll
  for (int off = 16; off > 0; off >>= 1) s += __shfl_down(s, off, 32);
  if (lane == 0) out[row] = s;
}

// fp32 -> bf16 hi(trunc)/lo(RNE residual), K-chunk-tiled layout [kc][row][32] so a
// block's per-chunk tile is one contiguous 4KB region.
__global__ __launch_bounds__(256) void bf16split_kernel(const float* __restrict__ X,
                                                        unsigned short* __restrict__ hi,
                                                        unsigned short* __restrict__ lo,
                                                        int rows) {
  const size_t i = (size_t)blockIdx.x * 256 + threadIdx.x; // float4 index
  const int row = (int)(i >> 7);        // 128 float4 per row
  const int c4  = (int)(i & 127) * 4;
  float4 v = *(const float4*)(X + (size_t)row * D_DIM + c4);
  const int kc = c4 >> 5, cc = c4 & 31;
  const size_t o = (size_t)kc * rows * BK + (size_t)row * BK + cc;
  unsigned int u[4] = {__float_as_uint(v.x), __float_as_uint(v.y),
                       __float_as_uint(v.z), __float_as_uint(v.w)};
  float xf[4] = {v.x, v.y, v.z, v.w};
  float lf[4];
#pragma unroll
  for (int j = 0; j < 4; ++j) lf[j] = xf[j] - __uint_as_float(u[j] & 0xFFFF0000u);
  uint2 hp, lp;
  hp.x = pack_hi_trunc(u[0], u[1]);
  hp.y = pack_hi_trunc(u[2], u[3]);
  lp.x = pack_bf16_rn(lf[0], lf[1]);
  lp.y = pack_bf16_rn(lf[2], lf[3]);
  *(uint2*)(hi + o) = hp;
  *(uint2*)(lo + o) = lp;
}

// ---------------- Path A: pre-split bf16 operands, async staged ----------------

__global__ __launch_bounds__(NTHREADS) void l2dist_wmma_pre_kernel(
    const unsigned short* __restrict__ Qh, const unsigned short* __restrict__ Ql,
    const unsigned short* __restrict__ Sh, const unsigned short* __restrict__ Sl,
    const float* __restrict__ q2, const float* __restrict__ s2,
    float* __restrict__ out, int nrows, int mcols) {
  __shared__ unsigned short lds[2 * BUF_HALVES]; // 60 KB double buffered

  const int t = threadIdx.x, lane = t & 31, wave = t >> 5;
  const int wm = wave & 1, wn = wave >> 1;   // wave tile 32 rows x 64 cols
  const int rowBase = blockIdx.y * BM, colBase = blockIdx.x * BN;

  const size_t qChunkStride = (size_t)nrows * BK;
  const size_t sChunkStride = (size_t)mcols * BK;
  const size_t qTile = (size_t)rowBase * BK;
  const size_t sTile = (size_t)colBase * BK;

  v8f acc[2][4];
  {
    v8f z = {0.f, 0.f, 0.f, 0.f, 0.f, 0.f, 0.f, 0.f};
#pragma unroll
    for (int i = 0; i < 2; ++i)
#pragma unroll
      for (int j = 0; j < 4; ++j) acc[i][j] = z;
  }

#if HAVE_ASYNC_LDS
  auto issue_chunk = [&](int kc, unsigned short* buf) {
    // Q tiles: 64 rows x 4 segs(16B) = 256 segs; 128 threads -> 2 each (hi & lo).
#pragma unroll
    for (int i = 0; i < 2; ++i) {
      const int idx = i * NTHREADS + t;
      const size_t qOff = (size_t)kc * qChunkStride + qTile + (size_t)idx * 8;
      const unsigned int qDst = (idx >> 2) * LDST + (idx & 3) * 8;
      __builtin_amdgcn_global_load_async_to_lds_b128(to_global_v4i(Qh + qOff),
                                                     to_lds_v4i(buf + QH_OFF + qDst), 0, 0);
      __builtin_amdgcn_global_load_async_to_lds_b128(to_global_v4i(Ql + qOff),
                                                     to_lds_v4i(buf + QL_OFF + qDst), 0, 0);
    }
    // S tiles: 128 rows x 4 segs = 512 segs; 128 threads -> 4 each (hi & lo).
#pragma unroll
    for (int i = 0; i < 4; ++i) {
      const int idx = i * NTHREADS + t;
      const size_t sOff = (size_t)kc * sChunkStride + sTile + (size_t)idx * 8;
      const unsigned int sDst = (idx >> 2) * LDST + (idx & 3) * 8;
      __builtin_amdgcn_global_load_async_to_lds_b128(to_global_v4i(Sh + sOff),
                                                     to_lds_v4i(buf + SH_OFF + sDst), 0, 0);
      __builtin_amdgcn_global_load_async_to_lds_b128(to_global_v4i(Sl + sOff),
                                                     to_lds_v4i(buf + SL_OFF + sDst), 0, 0);
    }
  };

  issue_chunk(0, lds);
  wait_async0();
  __syncthreads();

  int buf = 0;
  for (int kc = 0; kc < NCHUNK; ++kc) {
    if (kc + 1 < NCHUNK) issue_chunk(kc + 1, lds + (buf ^ 1) * BUF_HALVES);
    compute_chunk(lds + buf * BUF_HALVES, wm, wn, lane, acc);
    if (kc + 1 < NCHUNK) {
      wait_async0();
      __syncthreads();
      buf ^= 1;
    }
  }
#else
  uint4 qr[4], sr[8];
  auto load_chunk = [&](int kc) {
#pragma unroll
    for (int i = 0; i < 2; ++i) {
      const int idx = i * NTHREADS + t;
      const size_t qOff = (size_t)kc * qChunkStride + qTile + (size_t)idx * 8;
      qr[2 * i]     = *(const uint4*)(Qh + qOff);
      qr[2 * i + 1] = *(const uint4*)(Ql + qOff);
    }
#pragma unroll
    for (int i = 0; i < 4; ++i) {
      const int idx = i * NTHREADS + t;
      const size_t sOff = (size_t)kc * sChunkStride + sTile + (size_t)idx * 8;
      sr[2 * i]     = *(const uint4*)(Sh + sOff);
      sr[2 * i + 1] = *(const uint4*)(Sl + sOff);
    }
  };
  auto store_chunk = [&](unsigned short* bufp) {
#pragma unroll
    for (int i = 0; i < 2; ++i) {
      const int idx = i * NTHREADS + t;
      const unsigned int qDst = (idx >> 2) * LDST + (idx & 3) * 8;
      *(uint4*)(bufp + QH_OFF + qDst) = qr[2 * i];
      *(uint4*)(bufp + QL_OFF + qDst) = qr[2 * i + 1];
    }
#pragma unroll
    for (int i = 0; i < 4; ++i) {
      const int idx = i * NTHREADS + t;
      const unsigned int sDst = (idx >> 2) * LDST + (idx & 3) * 8;
      *(uint4*)(bufp + SH_OFF + sDst) = sr[2 * i];
      *(uint4*)(bufp + SL_OFF + sDst) = sr[2 * i + 1];
    }
  };

  load_chunk(0);
  store_chunk(lds);
  __syncthreads();
  int buf = 0;
  for (int kc = 0; kc < NCHUNK; ++kc) {
    if (kc + 1 < NCHUNK) load_chunk(kc + 1);
    compute_chunk(lds + buf * BUF_HALVES, wm, wn, lane, acc);
    if (kc + 1 < NCHUNK) {
      store_chunk(lds + (buf ^ 1) * BUF_HALVES);
      __syncthreads();
      buf ^= 1;
    }
  }
#endif

  epilogue(acc, q2, s2, out, rowBase, colBase, wm, wn, lane, mcols);
}

// ---------------- Path B: in-kernel conversion (small workspace) ----------------

static __device__ __forceinline__ void stage4(unsigned short* hb, unsigned short* lb,
                                              int row, int colv, float4 v) {
  unsigned int u[4] = {__float_as_uint(v.x), __float_as_uint(v.y),
                       __float_as_uint(v.z), __float_as_uint(v.w)};
  float xf[4] = {v.x, v.y, v.z, v.w};
  float lf[4];
#pragma unroll
  for (int j = 0; j < 4; ++j) lf[j] = xf[j] - __uint_as_float(u[j] & 0xFFFF0000u);
  unsigned int* hp = (unsigned int*)(hb + row * LDST + colv * 4);
  hp[0] = pack_hi_trunc(u[0], u[1]);
  hp[1] = pack_hi_trunc(u[2], u[3]);
  unsigned int* lp = (unsigned int*)(lb + row * LDST + colv * 4);
  lp[0] = pack_bf16_rn(lf[0], lf[1]);
  lp[1] = pack_bf16_rn(lf[2], lf[3]);
}

__global__ __launch_bounds__(NTHREADS) void l2dist_wmma_conv_kernel(
    const float* __restrict__ Q, const float* __restrict__ S,
    const float* __restrict__ q2, const float* __restrict__ s2,
    float* __restrict__ out, int mcols) {
  __shared__ unsigned short lds[2 * BUF_HALVES];

  const int t = threadIdx.x, lane = t & 31, wave = t >> 5;
  const int wm = wave & 1, wn = wave >> 1;
  const int rowBase = blockIdx.y * BM, colBase = blockIdx.x * BN;

  v8f acc[2][4];
  {
    v8f z = {0.f, 0.f, 0.f, 0.f, 0.f, 0.f, 0.f, 0.f};
#pragma unroll
    for (int i = 0; i < 2; ++i)
#pragma unroll
      for (int j = 0; j < 4; ++j) acc[i][j] = z;
  }

  float4 qreg[4], sreg[8];
  auto load_chunk = [&](int ko) {
    // Q: 64 rows x 8 float4 = 512; 128 threads -> 4 each.
#pragma unroll
    for (int i = 0; i < 4; ++i) {
      int idx = i * NTHREADS + t, r = idx >> 3, c = idx & 7;
      qreg[i] = *(const float4*)(Q + (size_t)(rowBase + r) * D_DIM + ko + c * 4);
    }
    // S: 128 rows x 8 float4 = 1024; 128 threads -> 8 each.
#pragma unroll
    for (int i = 0; i < 8; ++i) {
      int idx = i * NTHREADS + t, r = idx >> 3, c = idx & 7;
      sreg[i] = *(const float4*)(S + (size_t)(colBase + r) * D_DIM + ko + c * 4);
    }
  };
  auto store_chunk = [&](unsigned short* nb) {
#pragma unroll
    for (int i = 0; i < 4; ++i) {
      int idx = i * NTHREADS + t, r = idx >> 3, c = idx & 7;
      stage4(nb + QH_OFF, nb + QL_OFF, r, c, qreg[i]);
    }
#pragma unroll
    for (int i = 0; i < 8; ++i) {
      int idx = i * NTHREADS + t, r = idx >> 3, c = idx & 7;
      stage4(nb + SH_OFF, nb + SL_OFF, r, c, sreg[i]);
    }
  };

  load_chunk(0);
  store_chunk(lds);
  __syncthreads();

  int buf = 0;
  for (int kc = 0; kc < NCHUNK; ++kc) {
    if (kc + 1 < NCHUNK) load_chunk((kc + 1) * BK);
    compute_chunk(lds + buf * BUF_HALVES, wm, wn, lane, acc);
    if (kc + 1 < NCHUNK) {
      store_chunk(lds + (buf ^ 1) * BUF_HALVES);
      __syncthreads();
      buf ^= 1;
    }
  }

  epilogue(acc, q2, s2, out, rowBase, colBase, wm, wn, lane, mcols);
}

// ---------------- host ----------------

extern "C" void kernel_launch(void* const* d_in, const int* in_sizes, int n_in,
                              void* d_out, int out_size, void* d_ws, size_t ws_size,
                              hipStream_t stream) {
  const float* Q = (const float*)d_in[0];
  const float* S = (const float*)d_in[1];
  float* out = (float*)d_out;
  const int n = in_sizes[0] / D_DIM; // 8192
  const int m = in_sizes[1] / D_DIM; // 8192

  float* q2 = (float*)d_ws;
  float* s2 = q2 + n;

  rowsumsq_kernel<<<n / 8, 256, 0, stream>>>(Q, q2);
  rowsumsq_kernel<<<m / 8, 256, 0, stream>>>(S, s2);

  dim3 grid(m / BN, n / BM);
  const size_t need = (size_t)(n + m) * sizeof(float)       // q2, s2
                    + (size_t)(n + m) * D_DIM * 2 * 2;      // hi+lo bf16 for Q and S
  if (ws_size >= need) {
    unsigned short* Qh = (unsigned short*)(s2 + m);
    unsigned short* Ql = Qh + (size_t)n * D_DIM;
    unsigned short* Sh = Ql + (size_t)n * D_DIM;
    unsigned short* Sl = Sh + (size_t)m * D_DIM;
    bf16split_kernel<<<n / 2, 256, 0, stream>>>(Q, Qh, Ql, n);
    bf16split_kernel<<<m / 2, 256, 0, stream>>>(S, Sh, Sl, m);
    l2dist_wmma_pre_kernel<<<grid, NTHREADS, 0, stream>>>(Qh, Ql, Sh, Sl, q2, s2, out, n, m);
  } else {
    l2dist_wmma_conv_kernel<<<grid, NTHREADS, 0, stream>>>(Q, S, q2, s2, out, m);
  }
}
